// MultiScaleSpectralInit_40716289966147
// MI455X (gfx1250) — matrix operand order repaired and checked by reading, not compiled, not run on hardware
//
#include <hip/hip_runtime.h>
#include <cstdint>

// ---------------------------------------------------------------------------
// MultiScaleSpectralInit for MI455X (gfx1250, wave32, WMMA + TDM)
// Gram matrices + final linear via split-bf16 V_WMMA_F32_16X16X32_BF16,
// tiles streamed global->LDS by the Tensor Data Mover (double-buffered).
// ---------------------------------------------------------------------------

#define KNN 20
#define TILE_M 32
#define TILE_N 64
#define TILE_K 32
#define LDSK 34   // padded K stride (32 data + 2 pad dwords via TDM pad)

typedef __bf16 bf16;
typedef __attribute__((ext_vector_type(16))) __bf16 bf16x16;
typedef __attribute__((ext_vector_type(8)))  float  floatx8;

#if defined(__has_builtin)
#if __has_builtin(__builtin_amdgcn_tensor_load_to_lds) && \
    __has_builtin(__builtin_amdgcn_s_wait_tensorcnt)
#define HAVE_TDM 1
#endif
#endif
#ifndef HAVE_TDM
#define HAVE_TDM 0
#endif

#if HAVE_TDM
typedef __attribute__((ext_vector_type(4))) unsigned int uint32x4;
typedef __attribute__((ext_vector_type(8))) int int32x8;
typedef __attribute__((ext_vector_type(4))) int int32x4;

// Issue one 2-D TDM tile load: rows x 32 f32 from row-major [.,K] into LDS,
// padding +2 dwords per 32-dword row (LDS row stride = 34 floats).
__device__ __forceinline__ void tdm_load_tile(uint32_t lds_off, const float* gaddr,
                                              uint32_t rows, uint32_t K) {
  const uint64_t ga = (uint64_t)(uintptr_t)gaddr;
  uint32x4 g0;
  g0[0] = 1u;                                        // count=1, user descriptor
  g0[1] = lds_off;                                   // lds_addr (bytes)
  g0[2] = (uint32_t)(ga & 0xFFFFFFFFu);              // global_addr[31:0]
  g0[3] = (uint32_t)((ga >> 32) & 0x01FFFFFFu) |     // global_addr[56:32]
          (2u << 30);                                // type = 2 ("image")
  const uint32_t td0 = 1u << 30, td1 = 1u << 30;     // huge dims: no OOB clip
  int32x8 g1;
  g1[0] = (int)((2u << 16) |                         // data_size = 4 bytes
                (1u << 20) |                         // pad_enable
                (4u << 22) |                         // pad_interval: 32 dwords
                (1u << 25));                         // pad_amount: 2 dwords
  g1[1] = (int)((td0 & 0xFFFFu) << 16);              // tensor_dim0 lo (ab_addr=0)
  g1[2] = (int)((td0 >> 16) | ((td1 & 0xFFFFu) << 16));
  g1[3] = (int)((td1 >> 16) | ((uint32_t)TILE_K << 16));  // tile_dim0 = 32
  g1[4] = (int)rows;                                 // tile_dim1 = rows, dim2 = 0
  g1[5] = (int)K;                                    // tensor_dim0_stride lo32
  g1[6] = 0;
  g1[7] = 0;
  const int32x4 z4 = {0, 0, 0, 0};
  const int32x8 z8 = {0, 0, 0, 0, 0, 0, 0, 0};
  __builtin_amdgcn_tensor_load_to_lds(g0, g1, z4, z4, z8, 0);
}
#endif

// ------------------------------ helpers -----------------------------------

__device__ __forceinline__ float block_reduce_sum(float v, float* red) {
  #pragma unroll
  for (int off = 16; off; off >>= 1) v += __shfl_xor(v, off, 32);
  const int lane = threadIdx.x & 31;
  const int wave = threadIdx.x >> 5;
  if (lane == 0) red[wave] = v;
  __syncthreads();
  float r = (threadIdx.x < (blockDim.x >> 5)) ? red[threadIdx.x] : 0.0f;
  if (wave == 0) {
    #pragma unroll
    for (int off = 16; off; off >>= 1) r += __shfl_xor(r, off, 32);
    if (lane == 0) red[0] = r;
  }
  __syncthreads();
  float out = red[0];
  __syncthreads();
  return out;
}

__device__ __forceinline__ uint32_t pcg_hash(uint32_t x) {
  x = x * 747796405u + 2891336453u;
  uint32_t w = ((x >> ((x >> 28u) + 4u)) ^ x) * 277803737u;
  return (w >> 22u) ^ w;
}

__device__ __forceinline__ float rand_normal(uint32_t seed) {
  uint32_t a = pcg_hash(seed);
  uint32_t b = pcg_hash(seed ^ 0x9E3779B9u);
  float u1 = (float)((a >> 8) + 1u) * (1.0f / 16777216.0f);   // (0,1]
  float u2 = (float)(b >> 8) * (1.0f / 16777216.0f);          // [0,1)
  return sqrtf(-2.0f * logf(u1)) * cosf(6.28318530718f * u2);
}

// --------------------- row squared-norms: sq[b,n] = sum f^2 ----------------

__global__ __launch_bounds__(256) void rowsq_kernel(const float* __restrict__ F,
                                                    float* __restrict__ sq, int D) {
  __shared__ float red[32];
  const size_t row = blockIdx.x;
  const float* x = F + row * (size_t)D;
  float s = 0.0f;
  for (int d = threadIdx.x; d < D; d += 256) { float v = x[d]; s += v * v; }
  s = block_reduce_sum(s, red);
  if (threadIdx.x == 0) sq[row] = s;
}

// --------------------- split-bf16 NT GEMM with WMMA + TDM ------------------
// OUT[i,j] = sum_k X[i,k] * Y[j,k]  (per batch), fp32-ish via hi/lo bf16 split
// MODE 0: dist epilogue  out = max(sq[i] + sq[j] - 2*acc, 0)
// MODE 1: bias epilogue  out = acc + aux[j]

template <int MODE>
__global__ __launch_bounds__(256) void gemm_nt_kernel(
    const float* __restrict__ Xg, const float* __restrict__ Yg,
    float* __restrict__ Og, const float* __restrict__ auxg,
    int K, int Ncol,
    long long sX, long long sY, long long sO, long long sAux) {
  const int b = blockIdx.x;
  const float* X = Xg + (size_t)b * sX;
  const float* Y = Yg + (size_t)b * sY;
  float* O = Og + (size_t)b * sO;
  const float* aux = auxg + (size_t)b * sAux;

  const int i0 = blockIdx.y * TILE_M;
  const int j0 = blockIdx.z * TILE_N;

  const int tid = threadIdx.x;
  const int lane = tid & 31;
  const int wave = tid >> 5;
  const int waveRow = (wave & 1) << 4;    // 0,16
  const int waveCol = (wave >> 1) << 4;   // 0,16,32,48

  floatx8 acc;
  #pragma unroll
  for (int v = 0; v < 8; ++v) acc[v] = 0.0f;

#if HAVE_TDM
  // f32 tiles streamed by the Tensor Data Mover, double-buffered.
  __shared__ float Xf[2][TILE_M][LDSK];
  __shared__ float Yf[2][TILE_N][LDSK];

  // prologue: chunk 0 into buffer 0 (wave 0 issues; EXEC-independent DMA)
  if (tid < 32) {
    tdm_load_tile((uint32_t)(uintptr_t)&Xf[0][0][0], X + (size_t)i0 * K,
                  TILE_M, (uint32_t)K);
    tdm_load_tile((uint32_t)(uintptr_t)&Yf[0][0][0], Y + (size_t)j0 * K,
                  TILE_N, (uint32_t)K);
  }

  int cur = 0;
  for (int k0 = 0; k0 < K; k0 += TILE_K) {
    const bool more = (k0 + TILE_K) < K;
    if (more && tid < 32) {                       // prefetch next chunk via TDM
      tdm_load_tile((uint32_t)(uintptr_t)&Xf[cur ^ 1][0][0],
                    X + (size_t)i0 * K + (k0 + TILE_K), TILE_M, (uint32_t)K);
      tdm_load_tile((uint32_t)(uintptr_t)&Yf[cur ^ 1][0][0],
                    Y + (size_t)j0 * K + (k0 + TILE_K), TILE_N, (uint32_t)K);
    }
    // TDM ops of a wave retire in order: leaving the 2 newest outstanding
    // guarantees the current buffer's 2 loads are complete.
    if (more) __builtin_amdgcn_s_wait_tensorcnt(2);
    else      __builtin_amdgcn_s_wait_tensorcnt(0);
    __syncthreads();

    // fragments: split f32 -> bf16 hi/lo at build time (contiguous k = 16*(lane/16) + e)
    bf16x16 ahi, alo, bhi, blo;
    const int arow = waveRow + (lane & 15);
    const int akb  = (lane < 16) ? 0 : 16;
    #pragma unroll
    for (int e = 0; e < 16; ++e) {
      int kk = akb + e;
      float f = Xf[cur][arow][kk];
      bf16 h = (bf16)f;
      ahi[e] = h;
      alo[e] = (bf16)(f - (float)h);
    }
    const int bcol = waveCol + (lane & 15);
    const int bkb  = (lane < 16) ? 0 : 16;
    #pragma unroll
    for (int e = 0; e < 16; ++e) {
      float f = Yf[cur][bcol][bkb + e];
      bf16 h = (bf16)f;
      bhi[e] = h;
      blo[e] = (bf16)(f - (float)h);
    }

    // G = hi*hi + hi*lo + lo*hi  (drop lo*lo; ~fp32 accuracy)
    acc = __builtin_amdgcn_wmma_f32_16x16x32_bf16(false, ahi, false, bhi,
                                                  (short)0, acc, false, false);
    acc = __builtin_amdgcn_wmma_f32_16x16x32_bf16(false, ahi, false, blo,
                                                  (short)0, acc, false, false);
    acc = __builtin_amdgcn_wmma_f32_16x16x32_bf16(false, alo, false, bhi,
                                                  (short)0, acc, false, false);
    __syncthreads();   // everyone done reading buf[cur] before TDM reuses it
    cur ^= 1;
  }
#else
  // fallback: manual staging with split conversion in LDS
  __shared__ bf16 Xhi[TILE_M][LDSK];
  __shared__ bf16 Xlo[TILE_M][LDSK];
  __shared__ bf16 Yhi[TILE_N][LDSK];
  __shared__ bf16 Ylo[TILE_N][LDSK];

  for (int k0 = 0; k0 < K; k0 += TILE_K) {
    if (k0 + TILE_K < K) {
      __builtin_prefetch(&X[(size_t)(i0 + (tid & 31)) * K + k0 + TILE_K], 0, 0);
      __builtin_prefetch(&Y[(size_t)(j0 + (tid & 63)) * K + k0 + TILE_K], 0, 0);
    }
    for (int p = tid; p < TILE_M * TILE_K; p += 256) {
      int r = p >> 5, c = p & 31;
      float f = X[(size_t)(i0 + r) * K + (k0 + c)];
      bf16 h = (bf16)f;
      Xhi[r][c] = h;
      Xlo[r][c] = (bf16)(f - (float)h);
    }
    for (int p = tid; p < TILE_N * TILE_K; p += 256) {
      int r = p >> 5, c = p & 31;
      float f = Y[(size_t)(j0 + r) * K + (k0 + c)];
      bf16 h = (bf16)f;
      Yhi[r][c] = h;
      Ylo[r][c] = (bf16)(f - (float)h);
    }
    __syncthreads();

    bf16x16 ahi, alo, bhi, blo;
    const int arow = waveRow + (lane & 15);
    const int akb  = (lane < 16) ? 0 : 16;
    #pragma unroll
    for (int e = 0; e < 16; ++e) {
      int kk = akb + e;
      ahi[e] = Xhi[arow][kk];
      alo[e] = Xlo[arow][kk];
    }
    const int bcol = waveCol + (lane & 15);
    const int bkb  = (lane < 16) ? 0 : 16;
    #pragma unroll
    for (int e = 0; e < 16; ++e) {
      bhi[e] = Yhi[bcol][bkb + e];
      blo[e] = Ylo[bcol][bkb + e];
    }

    acc = __builtin_amdgcn_wmma_f32_16x16x32_bf16(false, ahi, false, bhi,
                                                  (short)0, acc, false, false);
    acc = __builtin_amdgcn_wmma_f32_16x16x32_bf16(false, ahi, false, blo,
                                                  (short)0, acc, false, false);
    acc = __builtin_amdgcn_wmma_f32_16x16x32_bf16(false, alo, false, bhi,
                                                  (short)0, acc, false, false);
    __syncthreads();
  }
#endif

  // C/D layout: lane holds col n = lane%16; VGPR v holds row v (+8 for hi lanes)
  const int n = j0 + waveCol + (lane & 15);
  const int mbase = i0 + waveRow + ((lane < 16) ? 0 : 8);
  if (MODE == 0) {
    const float sqn = aux[n];
    #pragma unroll
    for (int v = 0; v < 8; ++v) {
      int m = mbase + v;
      float d = aux[m] + sqn - 2.0f * acc[v];
      O[(size_t)m * Ncol + n] = fmaxf(d, 0.0f);
    }
  } else {
    const float bias = aux[n];
    #pragma unroll
    for (int v = 0; v < 8; ++v) {
      int m = mbase + v;
      O[(size_t)m * Ncol + n] = acc[v] + bias;
    }
  }
}

// --------------------- top-(KNN+1) smallest distances per row --------------
// One wave per row; 21 selection rounds via wave shuffles (drop self = first).

__global__ __launch_bounds__(256) void topk_kernel(const float* __restrict__ dist,
                                                   int N, int* __restrict__ knn_idx,
                                                   float* __restrict__ knn_w) {
  const size_t row = (size_t)blockIdx.x * 8 + (threadIdx.x >> 5);
  const int lane = threadIdx.x & 31;
  const float* drow = dist + row * (size_t)N;

  float ld[KNN + 1];
  int li[KNN + 1];
  #pragma unroll
  for (int i = 0; i <= KNN; ++i) { ld[i] = 3.4e38f; li[i] = 0x7fffffff; }

  for (int m = lane; m < N; m += 32) {
    float d = drow[m];
    if (d < ld[KNN] || (d == ld[KNN] && m < li[KNN])) {
      int p = KNN;
      while (p > 0 && (ld[p - 1] > d || (ld[p - 1] == d && li[p - 1] > m))) {
        ld[p] = ld[p - 1]; li[p] = li[p - 1]; --p;
      }
      ld[p] = d; li[p] = m;
    }
  }

  int head = 0;
  for (int r = 0; r <= KNN; ++r) {
    float d = (head <= KNN) ? ld[head] : 3.4e38f;
    int idx = (head <= KNN) ? li[head] : 0x7fffffff;
    float bd = d; int bi = idx;
    #pragma unroll
    for (int off = 16; off; off >>= 1) {
      float od = __shfl_xor(bd, off, 32);
      int   oi = __shfl_xor(bi, off, 32);
      if (od < bd || (od == bd && (unsigned)oi < (unsigned)bi)) { bd = od; bi = oi; }
    }
    if (idx == bi) head++;
    if (r >= 1 && lane == 0) {                 // drop first (self)
      knn_idx[row * KNN + (r - 1)] = bi;
      knn_w[row * KNN + (r - 1)] = expf(-bd * 0.5f);  // SIGMA=1
    }
  }
}

// --------------------- zero / scatter-symmetrize / normalize ---------------

__global__ __launch_bounds__(256) void zero_kernel(float* __restrict__ p, long long total) {
  long long i = (long long)blockIdx.x * 256 + threadIdx.x;
  if (i < total) p[i] = 0.0f;
}

__global__ __launch_bounds__(256) void scatter_kernel(const int* __restrict__ knn_idx,
                                                      const float* __restrict__ knn_w,
                                                      float* __restrict__ A, int N,
                                                      long long total) {
  long long p = (long long)blockIdx.x * 256 + threadIdx.x;
  if (p >= total) return;
  long long row = p / KNN;          // b*N + n
  long long b = row / N;
  int n = (int)(row - b * N);
  int m = knn_idx[p];
  float w = 0.5f * knn_w[p];
  float* Ab = A + b * (long long)N * N;
  atomicAdd(&Ab[(long long)n * N + m], w);
  atomicAdd(&Ab[(long long)m * N + n], w);
}

__global__ __launch_bounds__(256) void degree_kernel(const float* __restrict__ A,
                                                     float* __restrict__ dis, int N) {
  __shared__ float red[32];
  const size_t row = blockIdx.x;                // b*N + n ; rows are contiguous
  const float* Ar = A + row * (size_t)N;
  float s = 0.0f;
  for (int m = threadIdx.x; m < N; m += 256) s += Ar[m];
  s = block_reduce_sum(s, red);
  if (threadIdx.x == 0) dis[row] = 1.0f / sqrtf(s + 1e-8f);   // (deg+1e-8)^-0.5
}

__global__ __launch_bounds__(256) void norm_kernel(float* __restrict__ A,
                                                   const float* __restrict__ dis,
                                                   int N, long long total) {
  long long p = (long long)blockIdx.x * 256 + threadIdx.x;
  if (p >= total) return;
  long long nn = (long long)N * N;
  long long b = p / nn;
  long long nm = p - b * nn;
  int n = (int)(nm / N), m = (int)(nm - (long long)n * N);
  float v = A[p] * dis[b * N + n] * dis[b * N + m];
  if (n == m) v += 1.0f;                         // M = I + D^-1/2 A D^-1/2
  A[p] = v;
}

// --------------------- power iteration (persistent per batch) --------------

__global__ __launch_bounds__(256) void power_kernel(const float* __restrict__ Mg,
                                                    int N, int* __restrict__ sel,
                                                    int scale_id) {
  __shared__ float v_s[1024];
  __shared__ float vn_s[1024];
  __shared__ float eig_s[4][1024];
  __shared__ float red[32];
  __shared__ float amax[256];
  __shared__ int   aidx[256];

  const int b = blockIdx.x;
  const int tid = threadIdx.x;
  const float* Mb = Mg + (size_t)b * N * N;

  for (int slot = 0; slot < 4; ++slot) {
    for (int n = tid; n < N; n += 256) {
      uint32_t seed = ((uint32_t)scale_id << 28) ^ ((uint32_t)slot << 24) ^
                      ((uint32_t)b << 16) ^ (uint32_t)n;
      v_s[n] = rand_normal(seed);
    }
    __syncthreads();
    float ss = 0.0f;
    for (int n = tid; n < N; n += 256) ss += v_s[n] * v_s[n];
    ss = block_reduce_sum(ss, red);
    float inv = 1.0f / (sqrtf(ss) + 1e-8f);
    for (int n = tid; n < N; n += 256) v_s[n] *= inv;
    __syncthreads();

    for (int it = 0; it < 50; ++it) {
      for (int r = tid; r < N; r += 256) {
        const float* Mr = Mb + (size_t)r * N;
        float acc = 0.0f;
        for (int m = 0; m < N; ++m) acc += Mr[m] * v_s[m];
        vn_s[r] = acc;
      }
      __syncthreads();
      for (int p = 0; p < slot; ++p) {          // Gram-Schmidt deflation
        float d = 0.0f;
        for (int n = tid; n < N; n += 256) d += eig_s[p][n] * vn_s[n];
        d = block_reduce_sum(d, red);
        for (int n = tid; n < N; n += 256) vn_s[n] -= d * eig_s[p][n];
        __syncthreads();
      }
      float s2 = 0.0f;
      for (int n = tid; n < N; n += 256) s2 += vn_s[n] * vn_s[n];
      s2 = block_reduce_sum(s2, red);
      float i2 = 1.0f / (sqrtf(s2) + 1e-8f);
      for (int n = tid; n < N; n += 256) v_s[n] = vn_s[n] * i2;
      __syncthreads();
    }
    for (int n = tid; n < N; n += 256) eig_s[slot][n] = v_s[n];

    // argmax |v| (first occurrence on ties)
    float bv = -1.0f; int bi = 0x7fffffff;
    for (int n = tid; n < N; n += 256) {
      float a = fabsf(v_s[n]);
      if (a > bv || (a == bv && n < bi)) { bv = a; bi = n; }
    }
    amax[tid] = bv; aidx[tid] = bi;
    __syncthreads();
    for (int s = 128; s; s >>= 1) {
      if (tid < s) {
        if (amax[tid + s] > amax[tid] ||
            (amax[tid + s] == amax[tid] && aidx[tid + s] < aidx[tid])) {
          amax[tid] = amax[tid + s]; aidx[tid] = aidx[tid + s];
        }
      }
      __syncthreads();
    }
    if (tid == 0) sel[b * 4 + slot] = aidx[0];
    __syncthreads();
  }
}

// --------------------- gather prototypes + LayerNorm -----------------------

__global__ __launch_bounds__(256) void gather_kernel(
    const float* __restrict__ f8, const float* __restrict__ f16,
    const float* __restrict__ f32, const int* __restrict__ sel0,
    const int* __restrict__ sel1, const int* __restrict__ sel2,
    float* __restrict__ slots) {
  long long p = (long long)blockIdx.x * 256 + threadIdx.x;  // < 32*12*768
  int d = (int)(p % 768);
  int sg = (int)((p / 768) % 12);
  int b = (int)(p / (768 * 12));
  int scale = sg >> 2, slot = sg & 3;
  const float* f; int N; const int* sel;
  if (scale == 0)      { f = f8;  N = 64;   sel = sel0; }
  else if (scale == 1) { f = f16; N = 256;  sel = sel1; }
  else                 { f = f32; N = 1024; sel = sel2; }
  int n = sel[b * 4 + slot];
  slots[p] = f[((size_t)b * N + n) * 768 + d];
}

__global__ __launch_bounds__(256) void ln_kernel(const float* __restrict__ slots,
                                                 const float* __restrict__ gamma,
                                                 const float* __restrict__ beta,
                                                 float* __restrict__ xln) {
  __shared__ float red[32];
  const size_t row = blockIdx.x;                 // 384 rows
  const float* x = slots + row * 768;
  float s = 0.0f, s2 = 0.0f;
  for (int d = threadIdx.x; d < 768; d += 256) { float v = x[d]; s += v; s2 += v * v; }
  s = block_reduce_sum(s, red);
  s2 = block_reduce_sum(s2, red);
  float mu = s * (1.0f / 768.0f);
  float var = s2 * (1.0f / 768.0f) - mu * mu;
  float inv = 1.0f / sqrtf(var + 1e-5f);
  for (int d = threadIdx.x; d < 768; d += 256)
    xln[row * 768 + d] = (x[d] - mu) * inv * gamma[d] + beta[d];
}

// ------------------------------ host launcher ------------------------------

extern "C" void kernel_launch(void* const* d_in, const int* in_sizes, int n_in,
                              void* d_out, int out_size, void* d_ws, size_t ws_size,
                              hipStream_t stream) {
  (void)in_sizes; (void)n_in; (void)out_size;
  const float* feats[3] = {(const float*)d_in[0], (const float*)d_in[1],
                           (const float*)d_in[2]};
  const float* ln_gamma = (const float*)d_in[3];
  const float* ln_beta  = (const float*)d_in[4];
  const float* lin_w    = (const float*)d_in[5];
  const float* lin_b    = (const float*)d_in[6];
  float* out = (float*)d_out;

  const int B = 32, D = 768;
  const int Ns[3] = {64, 256, 1024};

  // workspace partition (aligned 256B); worst case N=1024 buffers are reused
  size_t off = 0;
  auto take = [&](size_t bytes) -> void* {
    void* p = (char*)d_ws + off;
    off = (off + bytes + 255) & ~(size_t)255;
    return p;
  };
  float* dist  = (float*)take((size_t)B * 1024 * 1024 * 4);   // dist -> aff -> M
  float* sq    = (float*)take((size_t)B * 1024 * 4);          // sq, then d^-1/2
  int*   knn_i = (int*)  take((size_t)B * 1024 * KNN * 4);
  float* knn_w = (float*)take((size_t)B * 1024 * KNN * 4);
  int*   sel   = (int*)  take((size_t)3 * B * 4 * 4);
  float* slots = (float*)take((size_t)B * 12 * D * 4);
  float* xln   = (float*)take((size_t)B * 12 * D * 4);
  if (off > ws_size) return;  // workspace too small; nothing safe to do

  for (int s = 0; s < 3; ++s) {
    const int N = Ns[s];
    const float* F = feats[s];
    const long long total = (long long)B * N * N;

    rowsq_kernel<<<B * N, 256, 0, stream>>>(F, sq, D);

    gemm_nt_kernel<0><<<dim3(B, N / 32, N / 64), 256, 0, stream>>>(
        F, F, dist, sq, D, N,
        (long long)N * D, (long long)N * D, (long long)N * N, (long long)N);

    topk_kernel<<<(B * N) / 8, 256, 0, stream>>>(dist, N, knn_i, knn_w);

    zero_kernel<<<(unsigned)((total + 255) / 256), 256, 0, stream>>>(dist, total);

    const long long sc_total = (long long)B * N * KNN;
    scatter_kernel<<<(unsigned)((sc_total + 255) / 256), 256, 0, stream>>>(
        knn_i, knn_w, dist, N, sc_total);

    degree_kernel<<<B * N, 256, 0, stream>>>(dist, sq, N);

    norm_kernel<<<(unsigned)((total + 255) / 256), 256, 0, stream>>>(dist, sq, N,
                                                                     total);

    power_kernel<<<B, 256, 0, stream>>>(dist, N, sel + s * B * 4, s);
  }

  const long long g_total = (long long)B * 12 * D;
  gather_kernel<<<(unsigned)((g_total + 255) / 256), 256, 0, stream>>>(
      feats[0], feats[1], feats[2], sel, sel + B * 4, sel + 2 * B * 4, slots);

  ln_kernel<<<B * 12, 256, 0, stream>>>(slots, ln_gamma, ln_beta, xln);

  // out[384,768] = xln @ lin_w^T + lin_b  (NT GEMM, same WMMA kernel)
  gemm_nt_kernel<1><<<dim3(1, (B * 12) / 32, D / 64), 256, 0, stream>>>(
      xln, lin_w, out, lin_b, D, D, 0, 0, 0, 0);
}
